// Model_39676907883759
// MI455X (gfx1250) — matrix operand order repaired and checked by reading, not compiled, not run on hardware
//
#include <hip/hip_runtime.h>

// CDNA5 flash-attention fwd: bf16 WMMA 16x16x32, f32 accumulate.
// Scores transposed through LDS in f32 -> softmax in A-layout with ONE
// shfl_xor(16) per reduction; aP packed directly in registers.
// Double-buffered K/V staging pipeline. B=4, S=4096, D=64, fp32 in/out.

typedef __attribute__((ext_vector_type(16))) __bf16 v16bf;
typedef __attribute__((ext_vector_type(8)))  __bf16 v8bf;
typedef __attribute__((ext_vector_type(4)))  __bf16 v4bf;
typedef __attribute__((ext_vector_type(8)))  float  v8f;

#define SQ   4096
#define DH   64
#define KT   64                 // keys per tile
#define NTIL (SQ / KT)          // 64 tiles
#define SKS  72                 // sK row stride (bf16): 36 dwords, gcd(36,64)=4
#define VST  72                 // sVt row stride (d-major V)
#define SST  68                 // sS row stride (f32): 68 dwords, gcd(68,64)=4

#define INV_KEEP 1.0090817356f  // 1/0.991
#define DROP16   590u           // round(0.009 * 65536): u16 < 590 -> drop

__device__ __forceinline__ unsigned pcg_hash(unsigned x) {
    x = x * 747796405u + 2891336453u;
    unsigned w = ((x >> ((x >> 28u) + 4u)) ^ x) * 277803737u;
    return (w >> 22u) ^ w;
}

// 16-half fragment from two aligned 16B LDS chunks (-> ds_load_b128 x2)
__device__ __forceinline__ v16bf frag16(const __bf16* p0, const __bf16* p1) {
    v8bf a = *(const v8bf*)p0;
    v8bf b = *(const v8bf*)p1;
    v16bf r;
#pragma unroll
    for (int j = 0; j < 8; ++j) { r[j] = a[j]; r[j + 8] = b[j]; }
    return r;
}

__device__ __forceinline__ void cvt_store_tile(__bf16* __restrict__ dK,
                                               __bf16* __restrict__ dVt,
                                               const float4* kreg, const float4* vreg,
                                               int tid) {
#pragma unroll
    for (int i = 0; i < 8; ++i) {
        int f4 = tid + i * 128;
        int e  = f4 << 2;
        int r  = e >> 6;                 // key row
        int d  = e & 63;                 // d col (multiple of 4)
        v4bf kb;
        kb[0] = (__bf16)kreg[i].x; kb[1] = (__bf16)kreg[i].y;
        kb[2] = (__bf16)kreg[i].z; kb[3] = (__bf16)kreg[i].w;
        *(v4bf*)(dK + r * SKS + d) = kb;              // key-major K
        dVt[(d + 0) * VST + r] = (__bf16)vreg[i].x;   // d-major (transposed) V
        dVt[(d + 1) * VST + r] = (__bf16)vreg[i].y;
        dVt[(d + 2) * VST + r] = (__bf16)vreg[i].z;
        dVt[(d + 3) * VST + r] = (__bf16)vreg[i].w;
    }
}

__global__ __launch_bounds__(128) void fa_fwd_bf16_wmma(
    const float* __restrict__ Q, const float* __restrict__ K,
    const float* __restrict__ V, const float* __restrict__ Sdiv,
    float* __restrict__ Out)
{
    __shared__ __bf16 sK [2][KT * SKS];
    __shared__ __bf16 sVt[2][DH * VST];
    __shared__ float  sS [4][16 * SST];   // per-wave f32 score transpose
    __shared__ float  sAl[4][16];         // per-wave alpha / rowsum broadcast

    const int tid  = threadIdx.x;
    const int wave = tid >> 5;
    const int lane = tid & 31;
    const int m    = lane & 15;
    const int h    = lane >> 4;

    const int qtiles = SQ / 64;
    const int b      = blockIdx.x / qtiles;
    const int qtile  = blockIdx.x % qtiles;
    const int q0     = qtile * 64 + wave * 16;

    // fold 1/s and log2(e) into Q: scores come out in the exp2 domain
    const float qscale = 1.4426950408889634f / Sdiv[0];

    // ---- Q tile (16x64) as two 16x32 bf16 A-fragments ----
    const float* qbase = Q + ((size_t)b * SQ + q0) * DH + m * DH;
    v16bf aQ[2];
#pragma unroll
    for (int f = 0; f < 2; ++f) {
        const float* p = qbase + f * 32 + h * 8;
        float4 x0 = *(const float4*)(p);
        float4 x1 = *(const float4*)(p + 4);
        float4 y0 = *(const float4*)(p + 16);
        float4 y1 = *(const float4*)(p + 20);
        aQ[f][0]  = (__bf16)(x0.x * qscale); aQ[f][1]  = (__bf16)(x0.y * qscale);
        aQ[f][2]  = (__bf16)(x0.z * qscale); aQ[f][3]  = (__bf16)(x0.w * qscale);
        aQ[f][4]  = (__bf16)(x1.x * qscale); aQ[f][5]  = (__bf16)(x1.y * qscale);
        aQ[f][6]  = (__bf16)(x1.z * qscale); aQ[f][7]  = (__bf16)(x1.w * qscale);
        aQ[f][8]  = (__bf16)(y0.x * qscale); aQ[f][9]  = (__bf16)(y0.y * qscale);
        aQ[f][10] = (__bf16)(y0.z * qscale); aQ[f][11] = (__bf16)(y0.w * qscale);
        aQ[f][12] = (__bf16)(y1.x * qscale); aQ[f][13] = (__bf16)(y1.y * qscale);
        aQ[f][14] = (__bf16)(y1.z * qscale); aQ[f][15] = (__bf16)(y1.w * qscale);
    }

    v8f accO[4];
    float rm = -INFINITY, rl = 0.0f;      // per-lane row stats (row = m)
#pragma unroll
    for (int c = 0; c < 4; ++c)
#pragma unroll
        for (int r = 0; r < 8; ++r) accO[c][r] = 0.0f;

    const size_t kvbatch = (size_t)b * SQ * DH;
    const float4* Kb4 = (const float4*)(K + kvbatch);
    const float4* Vb4 = (const float4*)(V + kvbatch);

    float4 kreg[8], vreg[8];
#pragma unroll
    for (int i = 0; i < 8; ++i) { kreg[i] = Kb4[tid + i * 128]; vreg[i] = Vb4[tid + i * 128]; }
    cvt_store_tile(sK[0], sVt[0], kreg, vreg, tid);
    __syncthreads();

    float* sSw = sS[wave];
    const unsigned grow = (unsigned)(b * SQ + q0 + m);

    for (int t = 0; t < NTIL; ++t) {
        const int cur = t & 1;
        const int kt0 = t * KT;

        if (t + 1 < NTIL) {
            const float4* kg4 = Kb4 + (size_t)(t + 1) * (KT * DH / 4);
            const float4* vg4 = Vb4 + (size_t)(t + 1) * (KT * DH / 4);
#pragma unroll
            for (int i = 0; i < 8; ++i) { kreg[i] = kg4[tid + i * 128]; vreg[i] = vg4[tid + i * 128]; }
        }

        const __bf16* cK = sK[cur];
        const __bf16* cV = sVt[cur];

        // ---- S = (Q*qscale) * K^T : 4 col-tiles x 2 WMMAs ----
        v8f accS[4];
#pragma unroll
        for (int nt = 0; nt < 4; ++nt) {
            const __bf16* kb = cK + (nt * 16 + m) * SKS + h * 8;
            v16bf b0 = frag16(kb,      kb + 16);
            v16bf b1 = frag16(kb + 32, kb + 48);
            v8f c = {};
            c = __builtin_amdgcn_wmma_f32_16x16x32_bf16(false, aQ[0], false, b0, (short)0, c, false, false);
            c = __builtin_amdgcn_wmma_f32_16x16x32_bf16(false, aQ[1], false, b1, (short)0, c, false, false);
            accS[nt] = c;
        }

        // ---- transpose scores to A-layout via LDS (f32) ----
#pragma unroll
        for (int nt = 0; nt < 4; ++nt)
#pragma unroll
            for (int r = 0; r < 8; ++r)
                sSw[(h * 8 + r) * SST + nt * 16 + m] = accS[nt][r];
        asm volatile("s_wait_dscnt 0" ::: "memory");

        // lane now owns 32 columns of row m (other 32 in lane m^16)
        float sv[2][16];
#pragma unroll
        for (int f = 0; f < 2; ++f) {
            const float* pr = sSw + m * SST + f * 32 + h * 8;
            float4 a0 = *(const float4*)(pr);
            float4 a1 = *(const float4*)(pr + 4);
            float4 b0 = *(const float4*)(pr + 16);
            float4 b1 = *(const float4*)(pr + 20);
            sv[f][0]  = a0.x; sv[f][1]  = a0.y; sv[f][2]  = a0.z; sv[f][3]  = a0.w;
            sv[f][4]  = a1.x; sv[f][5]  = a1.y; sv[f][6]  = a1.z; sv[f][7]  = a1.w;
            sv[f][8]  = b0.x; sv[f][9]  = b0.y; sv[f][10] = b0.z; sv[f][11] = b0.w;
            sv[f][12] = b1.x; sv[f][13] = b1.y; sv[f][14] = b1.z; sv[f][15] = b1.w;
        }

        // ---- row max: local tree + ONE shfl ----
        float mx = sv[0][0];
#pragma unroll
        for (int f = 0; f < 2; ++f)
#pragma unroll
            for (int j = 0; j < 16; ++j) mx = fmaxf(mx, sv[f][j]);
        mx = fmaxf(mx, __shfl_xor(mx, 16, 32));
        float mnew  = fmaxf(rm, mx);
        float alpha = exp2f(rm - mnew);
        rm = mnew;
        if (h == 0) sAl[wave][m] = alpha;   // broadcast alpha to C-layout lanes

        // ---- exp2 + dropout, pack aP directly; accumulate raw row sum ----
        float rs = 0.0f;
        v16bf aP[2];
#pragma unroll
        for (int f = 0; f < 2; ++f) {
#pragma unroll
            for (int j = 0; j < 16; j += 2) {
                float e0 = exp2f(sv[f][j]     - mnew);
                float e1 = exp2f(sv[f][j + 1] - mnew);
                rs += e0 + e1;
                unsigned col = (unsigned)(kt0 + f * 32 + ((j >> 3) << 4) + h * 8 + (j & 7));
                unsigned rnd = pcg_hash((grow * (unsigned)SQ + col) >> 1);
                aP[f][j]     = (__bf16)(e0 * (((rnd & 0xffffu) >= DROP16) ? INV_KEEP : 0.0f));
                aP[f][j + 1] = (__bf16)(e1 * (((rnd >> 16)     >= DROP16) ? INV_KEEP : 0.0f));
            }
        }
        rs += __shfl_xor(rs, 16, 32);
        rl = rl * alpha + rs;

        asm volatile("s_wait_dscnt 0" ::: "memory");

        // ---- rescale O accumulators (C-layout) by alpha[row] ----
#pragma unroll
        for (int r = 0; r < 8; ++r) {
            float a = sAl[wave][h * 8 + r];   // broadcast read
#pragma unroll
            for (int c = 0; c < 4; ++c) accO[c][r] *= a;
        }

        // ---- O += P*V : 4 d-chunks x 2 WMMAs ----
#pragma unroll
        for (int c = 0; c < 4; ++c) {
            v8f acc = accO[c];
#pragma unroll
            for (int f = 0; f < 2; ++f) {
                const __bf16* vb = cV + (c * 16 + m) * VST + f * 32 + h * 8;
                v16bf bV = frag16(vb, vb + 16);
                acc = __builtin_amdgcn_wmma_f32_16x16x32_bf16(false, aP[f], false, bV, (short)0, acc, false, false);
            }
            accO[c] = acc;
        }

        if (t + 1 < NTIL) cvt_store_tile(sK[cur ^ 1], sVt[cur ^ 1], kreg, vreg, tid);
        __syncthreads();
    }

    // ---- broadcast row sums to C-layout, normalize, write O ----
    if (h == 0) sAl[wave][m] = rl;
    asm volatile("s_wait_dscnt 0" ::: "memory");

    float* obase = Out + ((size_t)b * SQ + q0) * DH;
#pragma unroll
    for (int r = 0; r < 8; ++r) {
        int row = h * 8 + r;
        float inv = 1.0f / sAl[wave][row];
#pragma unroll
        for (int c = 0; c < 4; ++c)
            obase[row * DH + c * 16 + m] = accO[c][r] * inv;
    }
}

extern "C" void kernel_launch(void* const* d_in, const int* in_sizes, int n_in,
                              void* d_out, int out_size, void* d_ws, size_t ws_size,
                              hipStream_t stream) {
    const float* Q = (const float*)d_in[0];
    const float* K = (const float*)d_in[1];
    const float* V = (const float*)d_in[2];
    const float* s = (const float*)d_in[3];
    float* out = (float*)d_out;
    (void)in_sizes; (void)n_in; (void)out_size; (void)d_ws; (void)ws_size;

    dim3 grid(4 * (SQ / 64));   // B * query-tiles = 256 workgroups
    dim3 block(128);            // 4 waves; each wave owns 16 query rows
    fa_fwd_bf16_wmma<<<grid, block, 0, stream>>>(Q, K, V, s, out);
}
